// model_60627758351008
// MI455X (gfx1250) — matrix operand order repaired and checked by reading, not compiled, not run on hardware
//
#include <hip/hip_runtime.h>
#include <hip/hip_bf16.h>
#include <math.h>

typedef __attribute__((ext_vector_type(16))) _Float16 v16h;
typedef __attribute__((ext_vector_type(8)))  float    v8f;
typedef __attribute__((ext_vector_type(4)))  int      v4i_;

#define TSET 50
#define DIM  128
#define KCENT 128
#define NEG_INF (-__builtin_inff())

#if defined(__gfx1250__) && __has_builtin(__builtin_amdgcn_global_load_async_to_lds_b128) && __has_builtin(__builtin_amdgcn_s_wait_asynccnt)
#define USE_ASYNC_LDS 1
typedef __attribute__((address_space(1))) v4i_* gptr128;
typedef __attribute__((address_space(3))) v4i_* lptr128;
#else
#define USE_ASYNC_LDS 0
#endif

// logvol(x) = log(softplus(x) + 1e-10), beta = 1
__device__ __forceinline__ float logvol(float x) {
    float sp = (x > 20.f) ? x : log1pf(expf(x));
    return logf(sp + 1e-10f);
}

// sum within 16-lane groups of a wave32 (all lanes in group get the sum)
__device__ __forceinline__ float grp16_sum(float v) {
    v += __shfl_xor(v, 1, 32);
    v += __shfl_xor(v, 2, 32);
    v += __shfl_xor(v, 4, 32);
    v += __shfl_xor(v, 8, 32);
    return v;
}

// sum across full wave32
__device__ __forceinline__ float wave_sum(float v) {
    v += __shfl_xor(v, 1, 32);
    v += __shfl_xor(v, 2, 32);
    v += __shfl_xor(v, 4, 32);
    v += __shfl_xor(v, 8, 32);
    v += __shfl_xor(v, 16, 32);
    return v;
}

// ---------------------------------------------------------------------------
// Kernel 1: attention pooling for both tables.
// One block (128 threads = 4 waves) per set. Scores via v_wmma_f32_16x16x32_f16.
// Staging of the gathered embedding rows uses GLOBAL_LOAD_ASYNC_TO_LDS_B128
// (ASYNCcnt-tracked) when available.
// ---------------------------------------------------------------------------
__global__ void pool_kernel(const int* __restrict__ S, const float* __restrict__ M,
                            const float* __restrict__ ce, const float* __restrict__ re,
                            const float* __restrict__ ca, const float* __restrict__ ra,
                            float* __restrict__ c_all, float* __restrict__ r_all) {
    __shared__ float s_ce[TSET][DIM];
    __shared__ float s_re[TSET][DIM];
    __shared__ float s_attC[DIM];
    __shared__ float s_attR[DIM];
    __shared__ float s_zero[DIM];
    __shared__ float s_scC[64];
    __shared__ float s_scR[64];
    __shared__ float s_wC[TSET];
    __shared__ float s_wR[TSET];
    __shared__ float s_M[TSET];

    const int n   = blockIdx.x;
    const int tid = threadIdx.x;

    if (tid < TSET) s_M[tid] = M[n * TSET + tid];
    s_attC[tid] = ca[tid];
    s_attR[tid] = ra[tid];
    s_zero[tid] = 0.f;

#if USE_ASYNC_LDS
    // async gather: 4 floats (b128) per issue, tracked by ASYNCcnt
    for (int idx = tid; idx < TSET * 32; idx += 128) {
        int t  = idx >> 5;
        int d4 = (idx & 31) << 2;
        long long item = (long long)S[n * TSET + t];
        __builtin_amdgcn_global_load_async_to_lds_b128(
            (gptr128)(ce + item * DIM + d4),
            (lptr128)&s_ce[t][d4], 0, 0);
        __builtin_amdgcn_global_load_async_to_lds_b128(
            (gptr128)(re + item * DIM + d4),
            (lptr128)&s_re[t][d4], 0, 0);
    }
    __builtin_amdgcn_s_wait_asynccnt(0);
#else
    for (int idx = tid; idx < TSET * DIM; idx += 128) {
        int t = idx >> 7, d = idx & 127;
        long long item = (long long)S[n * TSET + t];
        s_ce[t][d] = ce[item * DIM + d];
        s_re[t][d] = re[item * DIM + d];
    }
#endif
    __syncthreads();

    // ---- scores via WMMA: per wave, tile of 16 t-rows; B col0=ca, col1=ra ----
    const int lane = tid & 31;
    const int wv   = tid >> 5;
    const int t0   = wv * 16;
    const int rowM = lane & 15;          // A: M = lane%16
    const int kHi  = (lane >> 4) * 8;    // A: K offset for upper half-wave
    const int tA   = t0 + rowM;

    // uniform-per-lane source pointers: no per-element conditionals
    const float* arowC = (tA < TSET) ? &s_ce[tA][0] : s_zero;
    const float* arowR = (tA < TSET) ? &s_re[tA][0] : s_zero;
    const int Nb = lane & 15;
    const float* brow = (Nb == 0) ? s_attC : ((Nb == 1) ? s_attR : s_zero);
    const int bOff = (lane >> 4) << 4;   // B: K = same table order as A (kHi + p, 16 + kHi + p)

    v8f accC = {};
    v8f accR = {};
    for (int kb = 0; kb < DIM; kb += 32) {
        v16h a_c, a_r, bmat;
#pragma unroll
        for (int p = 0; p < 8; ++p) {
            a_c[p]     = (_Float16)arowC[kb + kHi + p];
            a_c[p + 8] = (_Float16)arowC[kb + 16 + kHi + p];
            a_r[p]     = (_Float16)arowR[kb + kHi + p];
            a_r[p + 8] = (_Float16)arowR[kb + 16 + kHi + p];
        }
#pragma unroll
        for (int p = 0; p < 8; ++p) { bmat[p] = (_Float16)brow[kb + kHi + p]; bmat[p + 8] = (_Float16)brow[kb + 16 + kHi + p]; }
        accC = __builtin_amdgcn_wmma_f32_16x16x32_f16(false, a_c, false, bmat,
                                                      (short)0, accC, false, false);
        accR = __builtin_amdgcn_wmma_f32_16x16x32_f16(false, a_r, false, bmat,
                                                      (short)0, accR, false, false);
    }
    // D layout: element (M,N): vgpr r -> M = r + 8*(lane/16), N = lane%16
    {
        int N = lane & 15;
        int mBase = (lane >> 4) * 8;
        if (N == 0) {
#pragma unroll
            for (int r2 = 0; r2 < 8; ++r2) s_scC[t0 + mBase + r2] = accC[r2];
        }
        if (N == 1) {
#pragma unroll
            for (int r2 = 0; r2 < 8; ++r2) s_scR[t0 + mBase + r2] = accR[r2];
        }
    }
    __syncthreads();

    // ---- masked softmax over T (computed redundantly per thread; tiny) ----
    float mC = NEG_INF, mR = NEG_INF;
    for (int t = 0; t < TSET; ++t) {
        float msk = s_M[t];
        float sc = (msk > 0.f) ? s_scC[t] : NEG_INF;
        float sr = (msk > 0.f) ? s_scR[t] : NEG_INF;
        mC = fmaxf(mC, sc);
        mR = fmaxf(mR, sr);
    }
    float sumC = 0.f, sumR = 0.f;
    for (int t = 0; t < TSET; ++t) {
        float msk = s_M[t];
        sumC += (msk > 0.f) ? expf(s_scC[t] - mC) : 0.f;
        sumR += (msk > 0.f) ? expf(s_scR[t] - mR) : 0.f;
    }
    if (tid < TSET) {
        float msk = s_M[tid];
        s_wC[tid] = (msk > 0.f) ? (expf(s_scC[tid] - mC) / sumC) : 0.f;
        s_wR[tid] = (msk > 0.f) ? (expf(s_scR[tid] - mR) / sumR) : 0.f;
    }
    __syncthreads();

    // ---- pooled weighted sum in f32 ----
    float accc = 0.f, accr = 0.f;
    for (int t = 0; t < TSET; ++t) {
        accc += s_wC[t] * s_ce[t][tid];
        accr += s_wR[t] * s_re[t][tid];
    }
    c_all[(long long)n * DIM + tid] = accc;
    r_all[(long long)n * DIM + tid] = accr;
}

// ---------------------------------------------------------------------------
// Kernel 2: quantize. Forward of diff_softmax is hard argmax over K, and the
// global shift Z cancels in overlap = 0.5*(exp(Vi-Vq)+exp(Vi-Vc)).
// One block (128 threads) per set; subspace d = tid/16; shfl group reductions.
// ---------------------------------------------------------------------------
__global__ void quant_kernel(const float* __restrict__ c_all, const float* __restrict__ r_all,
                             const float* __restrict__ cc, const float* __restrict__ rc,
                             float* __restrict__ cq_all, float* __restrict__ rq_all) {
    const int n = blockIdx.x;
    const int j = threadIdx.x;                 // dim 0..127
    const long long base = (long long)n * DIM + j;

    float cj = c_all[base], rj = r_all[base];
    float mq = cj - rj, Mq = cj + rj;
    float Vq = grp16_sum(logvol(2.f * rj));    // [per subspace d = j/16]

    float best = NEG_INF;
    int bestk = 0;
    for (int k = 0; k < KCENT; ++k) {
        if (k + 8 < KCENT) {
            __builtin_prefetch(&cc[(k + 8) * DIM + j], 0, 1);
            __builtin_prefetch(&rc[(k + 8) * DIM + j], 0, 1);
        }
        float ck = cc[k * DIM + j];
        float rk = rc[k * DIM + j];
        float lo = fmaxf(mq, ck - rk);
        float hi = fminf(Mq, ck + rk);
        float Vi = grp16_sum(logvol(hi - lo));
        float Vc = grp16_sum(logvol(2.f * rk));
        float ov = 0.5f * (expf(Vi - Vq) + expf(Vi - Vc));
        if (ov > best) { best = ov; bestk = k; }   // strict > : first argmax, like jnp
    }
    cq_all[base] = cc[bestk * DIM + j];
    rq_all[base] = rc[bestk * DIM + j];
}

// ---------------------------------------------------------------------------
// Kernel 3a: 26 intersection log-volumes per instance + per-instance single-max.
// One wave per instance; lane covers 4 dims.
// ---------------------------------------------------------------------------
__device__ __forceinline__ float vol1(const float* am, const float* aM) {
    float a = 0.f;
#pragma unroll
    for (int t = 0; t < 4; ++t) a += logvol(aM[t] - am[t]);
    return wave_sum(a);
}
__device__ __forceinline__ float vol2(const float* am, const float* aM,
                                      const float* bm, const float* bM) {
    float a = 0.f;
#pragma unroll
    for (int t = 0; t < 4; ++t)
        a += logvol(fminf(aM[t], bM[t]) - fmaxf(am[t], bm[t]));
    return wave_sum(a);
}
__device__ __forceinline__ float vol3(const float* am, const float* aM,
                                      const float* bm, const float* bM,
                                      const float* cm, const float* cM) {
    float a = 0.f;
#pragma unroll
    for (int t = 0; t < 4; ++t)
        a += logvol(fminf(fminf(aM[t], bM[t]), cM[t]) - fmaxf(fmaxf(am[t], bm[t]), cm[t]));
    return wave_sum(a);
}

__global__ void boxes_kernel(const int* __restrict__ instances,
                             const float* __restrict__ c_all, const float* __restrict__ r_all,
                             const float* __restrict__ cq_all, const float* __restrict__ rq_all,
                             float* __restrict__ Cbuf, float* __restrict__ smax, int B) {
    const int wv = threadIdx.x >> 5;
    const int lane = threadIdx.x & 31;
    const int b = blockIdx.x * 4 + wv;
    if (b >= B) return;

    int si[3];
    si[0] = instances[b * 3 + 0];
    si[1] = instances[b * 3 + 1];
    si[2] = instances[b * 3 + 2];

    float bm[3][4], bM[3][4], qm[3][4], qM[3][4];
#pragma unroll
    for (int s = 0; s < 3; ++s) {
#pragma unroll
        for (int t = 0; t < 4; ++t) {
            int jj = lane + 32 * t;
            long long base = (long long)si[s] * DIM + jj;
            float c = c_all[base], r = r_all[base];
            float cq = cq_all[base], rq = rq_all[base];
            bm[s][t] = c - r;  bM[s][t] = c + r;
            qm[s][t] = cq - rq; qM[s][t] = cq + rq;
        }
    }

    float v[26];
    // singles
    v[0] = vol1(bm[0], bM[0]);
    v[1] = vol1(bm[1], bM[1]);
    v[2] = vol1(bm[2], bM[2]);
    v[3] = vol1(qm[0], qM[0]);
    v[4] = vol1(qm[1], qM[1]);
    v[5] = vol1(qm[2], qM[2]);
    // pairs
    v[6]  = vol2(bm[0], bM[0], bm[1], bM[1]);
    v[7]  = vol2(bm[0], bM[0], qm[1], qM[1]);
    v[8]  = vol2(qm[0], qM[0], bm[1], bM[1]);
    v[9]  = vol2(qm[0], qM[0], qm[1], qM[1]);
    v[10] = vol2(bm[1], bM[1], bm[2], bM[2]);
    v[11] = vol2(bm[1], bM[1], qm[2], qM[2]);
    v[12] = vol2(qm[1], qM[1], bm[2], bM[2]);
    v[13] = vol2(qm[1], qM[1], qm[2], qM[2]);
    v[14] = vol2(bm[2], bM[2], bm[0], bM[0]);
    v[15] = vol2(bm[2], bM[2], qm[0], qM[0]);
    v[16] = vol2(qm[2], qM[2], bm[0], bM[0]);
    v[17] = vol2(qm[2], qM[2], qm[0], qM[0]);
    // triples
    v[18] = vol3(bm[0], bM[0], bm[1], bM[1], bm[2], bM[2]);
    v[19] = vol3(bm[0], bM[0], bm[1], bM[1], qm[2], qM[2]);
    v[20] = vol3(bm[0], bM[0], qm[1], qM[1], bm[2], bM[2]);
    v[21] = vol3(qm[0], qM[0], bm[1], bM[1], bm[2], bM[2]);
    v[22] = vol3(bm[0], bM[0], qm[1], qM[1], qm[2], qM[2]);
    v[23] = vol3(qm[0], qM[0], bm[1], bM[1], qm[2], qM[2]);
    v[24] = vol3(qm[0], qM[0], qm[1], qM[1], bm[2], bM[2]);
    v[25] = vol3(qm[0], qM[0], qm[1], qM[1], qm[2], qM[2]);

    if (lane == 0) {
        float m6 = v[0];
#pragma unroll
        for (int c = 1; c < 6; ++c) m6 = fmaxf(m6, v[c]);
        smax[b] = m6;
#pragma unroll
        for (int c = 0; c < 26; ++c) Cbuf[(long long)c * B + b] = v[c];
    }
}

// ---------------------------------------------------------------------------
// Kernel 3b: global max over per-instance single-maxima (Z is batch-global).
// ---------------------------------------------------------------------------
__global__ void reduce_max_kernel(const float* __restrict__ smax, float* __restrict__ Z, int B) {
    __shared__ float sm[256];
    float m = NEG_INF;
    for (int i = threadIdx.x; i < B; i += 256) m = fmaxf(m, smax[i]);
    sm[threadIdx.x] = m;
    __syncthreads();
    for (int s = 128; s > 0; s >>= 1) {
        if (threadIdx.x < s) sm[threadIdx.x] = fmaxf(sm[threadIdx.x], sm[threadIdx.x + s]);
        __syncthreads();
    }
    if (threadIdx.x == 0) *Z = sm[0];
}

// ---------------------------------------------------------------------------
// Kernel 3c: out = exp(C - Z)
// ---------------------------------------------------------------------------
__global__ void expout_kernel(const float* __restrict__ Cbuf, const float* __restrict__ Z,
                              float* __restrict__ out, int total) {
    int i = blockIdx.x * 256 + threadIdx.x;
    if (i < total) out[i] = expf(Cbuf[i] - *Z);
}

// ---------------------------------------------------------------------------
extern "C" void kernel_launch(void* const* d_in, const int* in_sizes, int n_in,
                              void* d_out, int out_size, void* d_ws, size_t ws_size,
                              hipStream_t stream) {
    const int*   S   = (const int*)d_in[0];
    const float* M   = (const float*)d_in[1];
    const int*   ins = (const int*)d_in[2];
    // d_in[3] = overlaps : unused by the forward pass
    const float* ce  = (const float*)d_in[4];
    const float* re  = (const float*)d_in[5];
    const float* cc  = (const float*)d_in[6];
    const float* rc  = (const float*)d_in[7];
    const float* ca  = (const float*)d_in[8];
    const float* ra  = (const float*)d_in[9];
    float* out = (float*)d_out;

    const int NS = in_sizes[1] / TSET;   // number of sets (4096)
    const int B  = in_sizes[2] / 3;      // batch (8192)

    float* ws     = (float*)d_ws;
    float* c_all  = ws;
    float* r_all  = c_all + (size_t)NS * DIM;
    float* cq_all = r_all + (size_t)NS * DIM;
    float* rq_all = cq_all + (size_t)NS * DIM;
    float* Cbuf   = rq_all + (size_t)NS * DIM;
    float* smax   = Cbuf + (size_t)26 * B;
    float* Zbuf   = smax + (size_t)B;

    pool_kernel<<<NS, 128, 0, stream>>>(S, M, ce, re, ca, ra, c_all, r_all);
    quant_kernel<<<NS, 128, 0, stream>>>(c_all, r_all, cc, rc, cq_all, rq_all);
    boxes_kernel<<<(B + 3) / 4, 128, 0, stream>>>(ins, c_all, r_all, cq_all, rq_all,
                                                  Cbuf, smax, B);
    reduce_max_kernel<<<1, 256, 0, stream>>>(smax, Zbuf, B);
    expout_kernel<<<(26 * B + 255) / 256, 256, 0, stream>>>(Cbuf, Zbuf, out, 26 * B);
}
